// Layer_22282290331740
// MI455X (gfx1250) — compile-verified, matrix operand order checked by reading, not run on hardware
//
#include <hip/hip_runtime.h>
#include <math.h>

// ---------------- problem constants (from reference) ----------------
#define Bz 2
#define Sz 2048
#define Dz 1024
#define Hz 16
#define DHz 64
#define FFz 4096
#define NNz 1024
#define NPz 512
#define KNz 16
#define KPz 32
#define BSz (Bz*Sz)   // 4096

typedef _Float16 half_t;
typedef __attribute__((ext_vector_type(16))) _Float16 v16h;
typedef __attribute__((ext_vector_type(8)))  _Float16 v8h;
typedef __attribute__((ext_vector_type(8)))  float    v8f;

// =====================================================================
// convert f32 -> f16 (flat)
// =====================================================================
__global__ void k_cvt(const float* __restrict__ in, half_t* __restrict__ out, int n) {
    for (int i = blockIdx.x * blockDim.x + threadIdx.x; i < n; i += gridDim.x * blockDim.x)
        out[i] = (half_t)in[i];
}

// convert + transpose: in[R,C] f32 -> out[C,R] f16
__global__ void k_cvt_t(const float* __restrict__ in, half_t* __restrict__ out, int R, int C) {
    int n = R * C;
    for (int i = blockIdx.x * blockDim.x + threadIdx.x; i < n; i += gridDim.x * blockDim.x) {
        int c = i / R, r = i - c * R;
        out[i] = (half_t)in[(size_t)r * C + c];
    }
}

// =====================================================================
// layernorm over rows of length D; writes f32 and f16 copies
// =====================================================================
__global__ __launch_bounds__(256) void k_ln(const float* __restrict__ x,
                                            const float* __restrict__ g,
                                            const float* __restrict__ b,
                                            float* __restrict__ of,
                                            half_t* __restrict__ oh) {
    int row = blockIdx.x, tid = threadIdx.x;
    const float* xr = x + (size_t)row * Dz;
    __shared__ float red[256];
    float s = 0.f;
    for (int i = tid; i < Dz; i += 256) s += xr[i];
    red[tid] = s; __syncthreads();
    for (int st = 128; st > 0; st >>= 1) { if (tid < st) red[tid] += red[tid + st]; __syncthreads(); }
    float mu = red[0] * (1.0f / Dz);
    __syncthreads();
    float s2 = 0.f;
    for (int i = tid; i < Dz; i += 256) { float d = xr[i] - mu; s2 += d * d; }
    red[tid] = s2; __syncthreads();
    for (int st = 128; st > 0; st >>= 1) { if (tid < st) red[tid] += red[tid + st]; __syncthreads(); }
    float inv = rsqrtf(red[0] * (1.0f / Dz) + 1e-5f);
    for (int i = tid; i < Dz; i += 256) {
        float v = (xr[i] - mu) * inv * g[i] + b[i];
        of[(size_t)row * Dz + i] = v;
        oh[(size_t)row * Dz + i] = (half_t)v;
    }
}

// =====================================================================
// Register-blocked WMMA GEMM: C[M,N] = A[M,K] @ Wt[N,K]^T (+bias[N])
// A, Wt f16 row-major.  Each wave computes a 32x64 macro-tile:
//   2 M-tiles x 4 N-tiles = 8 accumulators; per k-step 2 A-tile + 4 B-tile
//   loads feed 8 WMMAs (vs 1 WMMA / 2 tile-loads unblocked) -> 2.7x better
//   L2 arithmetic intensity.  K%32==0, M%128==0, N%64==0.
// block = 128 threads = 4 waves stacked in M.
// =====================================================================
__global__ __launch_bounds__(128) void k_gemm(const half_t* __restrict__ A,
                                              const half_t* __restrict__ Wt,
                                              const float* __restrict__ bias,
                                              float* __restrict__ Cf,
                                              half_t* __restrict__ Ch,
                                              int M, int N, int K) {
    int lane = threadIdx.x & 31, wave = threadIdx.x >> 5;
    int n0 = blockIdx.x * 64;
    int m0 = (blockIdx.y * 4 + wave) * 32;
    if (m0 >= M) return;
    int l15 = lane & 15;
    int kb = (lane < 16) ? 0 : 8;        // K-offset per ISA 16-bit A/B striping
    const half_t* ap = A  + (size_t)(m0 + l15) * K + kb;
    const half_t* bp = Wt + (size_t)(n0 + l15) * K + kb;
    const size_t aStep = (size_t)16 * K;     // next M-tile row group
    const size_t bStep = (size_t)16 * K;     // next N-tile row group

    v8f acc[2][4];
#pragma unroll
    for (int mi = 0; mi < 2; mi++)
#pragma unroll
        for (int nj = 0; nj < 4; nj++) acc[mi][nj] = {};

    for (int k0 = 0; k0 < K; k0 += 32) {
        __builtin_prefetch(ap + k0 + 128, 0, 1);            // -> global_prefetch_b8
        __builtin_prefetch(bp + k0 + 128, 0, 1);
        v16h a[2], bm[4];
#pragma unroll
        for (int mi = 0; mi < 2; mi++) {
            v8h lo = *(const v8h*)(ap + mi * aStep + k0);
            v8h hi = *(const v8h*)(ap + mi * aStep + k0 + 16);
#pragma unroll
            for (int i = 0; i < 8; i++) { a[mi][i] = lo[i]; a[mi][i + 8] = hi[i]; }
        }
#pragma unroll
        for (int nj = 0; nj < 4; nj++) {
            v8h lo = *(const v8h*)(bp + nj * bStep + k0);
            v8h hi = *(const v8h*)(bp + nj * bStep + k0 + 16);
#pragma unroll
            for (int i = 0; i < 8; i++) { bm[nj][i] = lo[i]; bm[nj][i + 8] = hi[i]; }
        }
#pragma unroll
        for (int mi = 0; mi < 2; mi++)
#pragma unroll
            for (int nj = 0; nj < 4; nj++)
                acc[mi][nj] = __builtin_amdgcn_wmma_f32_16x16x32_f16(
                    false, a[mi], false, bm[nj], (short)0, acc[mi][nj], false, false);
    }

#pragma unroll
    for (int nj = 0; nj < 4; nj++) {
        int n = n0 + nj * 16 + l15;
        float bvv = bias ? bias[n] : 0.f;
#pragma unroll
        for (int mi = 0; mi < 2; mi++) {
#pragma unroll
            for (int r = 0; r < 8; r++) {
                int m = m0 + mi * 16 + r + ((lane < 16) ? 0 : 8);
                float v = acc[mi][nj][r] + bvv;
                if (Cf) Cf[(size_t)m * N + n] = v;
                if (Ch) Ch[(size_t)m * N + n] = (half_t)v;
            }
        }
    }
}

// =====================================================================
// transpose V: v[B,S, h*64+dh] f16 -> vT[(b*H+h)*64+dh][S] f16
// =====================================================================
__global__ void k_vt(const half_t* __restrict__ v, half_t* __restrict__ vt) {
    int n = Bz * Hz * DHz * Sz;
    for (int i = blockIdx.x * blockDim.x + threadIdx.x; i < n; i += gridDim.x * blockDim.x) {
        int s   = i & (Sz - 1);
        int dh  = (i >> 11) & (DHz - 1);
        int bh  = i >> 17;              // Sz*DHz = 2^17
        int b   = bh >> 4, h = bh & 15;
        vt[i] = v[((size_t)(b * Sz + s)) * Dz + h * DHz + dh];
    }
}

// =====================================================================
// flash attention: 1 wave per 16-query tile per (b,h); online softmax over
// 32-key blocks.  Q·K^T and P·V via WMMA; P transposed C->A layout via LDS.
// =====================================================================
__global__ __launch_bounds__(32) void k_attn(const half_t* __restrict__ Q,
                                             const half_t* __restrict__ Kh,
                                             const half_t* __restrict__ Vt,
                                             float* __restrict__ ctx_f,
                                             half_t* __restrict__ ctx_h) {
    int lane = threadIdx.x & 31;
    int bh = blockIdx.y, b = bh >> 4, h = bh & 15;
    int q0 = blockIdx.x * 16;
    int l15 = lane & 15;
    int kb  = (lane < 16) ? 0 : 8;
    int hi8 = (lane < 16) ? 0 : 8;

    const half_t* qrow = Q + ((size_t)(b * Sz + q0 + l15)) * Dz + h * DHz + kb;
    v16h qa0, qa1;
    {
        v8h lo = *(const v8h*)(qrow),      hi = *(const v8h*)(qrow + 16);
#pragma unroll
        for (int i = 0; i < 8; i++) { qa0[i] = lo[i]; qa0[i + 8] = hi[i]; }
        lo = *(const v8h*)(qrow + 32);     hi = *(const v8h*)(qrow + 48);
#pragma unroll
        for (int i = 0; i < 8; i++) { qa1[i] = lo[i]; qa1[i + 8] = hi[i]; }
    }

    v8f o[4]; o[0] = {}; o[1] = {}; o[2] = {}; o[3] = {};
    float mrow[8], lrow[8];
#pragma unroll
    for (int r = 0; r < 8; r++) { mrow[r] = -1e30f; lrow[r] = 0.f; }

    __shared__ float pbuf[16 * 32];

    for (int j0 = 0; j0 < Sz; j0 += 32) {
        v8f s0 = {}, s1 = {};
        {   // keys j0 .. j0+15
            const half_t* kr = Kh + ((size_t)(b * Sz + j0 + l15)) * Dz + h * DHz + kb;
            v16h b0, b1;
            v8h lo = *(const v8h*)kr,        hi = *(const v8h*)(kr + 16);
#pragma unroll
            for (int i = 0; i < 8; i++) { b0[i] = lo[i]; b0[i + 8] = hi[i]; }
            lo = *(const v8h*)(kr + 32);     hi = *(const v8h*)(kr + 48);
#pragma unroll
            for (int i = 0; i < 8; i++) { b1[i] = lo[i]; b1[i + 8] = hi[i]; }
            s0 = __builtin_amdgcn_wmma_f32_16x16x32_f16(false, qa0, false, b0, (short)0, s0, false, false);
            s0 = __builtin_amdgcn_wmma_f32_16x16x32_f16(false, qa1, false, b1, (short)0, s0, false, false);
        }
        {   // keys j0+16 .. j0+31
            const half_t* kr = Kh + ((size_t)(b * Sz + j0 + 16 + l15)) * Dz + h * DHz + kb;
            v16h b0, b1;
            v8h lo = *(const v8h*)kr,        hi = *(const v8h*)(kr + 16);
#pragma unroll
            for (int i = 0; i < 8; i++) { b0[i] = lo[i]; b0[i + 8] = hi[i]; }
            lo = *(const v8h*)(kr + 32);     hi = *(const v8h*)(kr + 48);
#pragma unroll
            for (int i = 0; i < 8; i++) { b1[i] = lo[i]; b1[i + 8] = hi[i]; }
            s1 = __builtin_amdgcn_wmma_f32_16x16x32_f16(false, qa0, false, b0, (short)0, s1, false, false);
            s1 = __builtin_amdgcn_wmma_f32_16x16x32_f16(false, qa1, false, b1, (short)0, s1, false, false);
        }
        // online softmax update (per C row: reduce over 16-lane groups)
#pragma unroll
        for (int r = 0; r < 8; r++) {
            float a0 = s0[r] * 0.125f, a1 = s1[r] * 0.125f;
            float mx = fmaxf(a0, a1);
#pragma unroll
            for (int off = 1; off < 16; off <<= 1) mx = fmaxf(mx, __shfl_xor(mx, off, 32));
            float mnew = fmaxf(mrow[r], mx);
            float corr = __expf(mrow[r] - mnew);
            mrow[r] = mnew;
            float p0 = __expf(a0 - mnew), p1 = __expf(a1 - mnew);
            s0[r] = p0; s1[r] = p1;
            float ps = p0 + p1;
#pragma unroll
            for (int off = 1; off < 16; off <<= 1) ps += __shfl_xor(ps, off, 32);
            lrow[r] = lrow[r] * corr + ps;
            o[0][r] *= corr; o[1][r] *= corr; o[2][r] *= corr; o[3][r] *= corr;
        }
        // transpose P (C layout -> A layout) through LDS
        __syncthreads();
#pragma unroll
        for (int r = 0; r < 8; r++) {
            int row = r + hi8;
            pbuf[row * 32 + l15]      = s0[r];
            pbuf[row * 32 + 16 + l15] = s1[r];
        }
        __syncthreads();
        v16h pa;
        {
            const float* pr = pbuf + l15 * 32 + kb;
#pragma unroll
            for (int i = 0; i < 8; i++) { pa[i] = (half_t)pr[i]; pa[i + 8] = (half_t)pr[16 + i]; }
        }
        // O += P @ V  (V pre-transposed: [bh][dh][S])
        const half_t* vbase = Vt + ((size_t)bh * DHz) * Sz + j0 + kb;
#pragma unroll
        for (int nt = 0; nt < 4; nt++) {
            const half_t* vr = vbase + (size_t)(nt * 16 + l15) * Sz;
            v8h lo = *(const v8h*)vr, hi = *(const v8h*)(vr + 16);
            v16h vb;
#pragma unroll
            for (int i = 0; i < 8; i++) { vb[i] = lo[i]; vb[i + 8] = hi[i]; }
            o[nt] = __builtin_amdgcn_wmma_f32_16x16x32_f16(false, pa, false, vb, (short)0, o[nt], false, false);
        }
    }
    // epilogue: normalize + store f32 and f16 context
#pragma unroll
    for (int r = 0; r < 8; r++) {
        float inv = 1.0f / lrow[r];
        int m = q0 + r + hi8;
        size_t base = ((size_t)(b * Sz + m)) * Dz + h * DHz;
#pragma unroll
        for (int nt = 0; nt < 4; nt++) {
            float v = o[nt][r] * inv;
            ctx_f[base + nt * 16 + l15] = v;
            ctx_h[base + nt * 16 + l15] = (half_t)v;
        }
    }
}

// =====================================================================
// router: pw softmax, combine scores, top-16, softmax weights, weighted
// neuron sum, residual, selection scatter, topk_idx output.  1 block/row.
// =====================================================================
__global__ __launch_bounds__(256) void k_router(const float* __restrict__ x,
                                                const float* __restrict__ nrm,
                                                const float* __restrict__ ctx,
                                                const float* __restrict__ tsc,
                                                const float* __restrict__ csc,
                                                const float* __restrict__ wpr,
                                                const float* __restrict__ bpr,
                                                const float* __restrict__ neurons,
                                                float* __restrict__ rout_f,
                                                half_t* __restrict__ rout_h,
                                                float* __restrict__ x1,
                                                int* __restrict__ topk_out,
                                                float* __restrict__ sel_out) {
    int row = blockIdx.x, tid = threadIdx.x;
    __shared__ float red[256];
    __shared__ float sc[NNz];
    __shared__ float rv[256];
    __shared__ int   ri[256];
    __shared__ int   bidx[KNz];
    __shared__ float bval[KNz];
    __shared__ float bw[2];
    __shared__ float wsh[KNz];

    const float* nr = nrm + (size_t)row * Dz;
    const float* cr = ctx + (size_t)row * Dz;
    // projection weights (2 logits)
    float d0 = 0.f, d1 = 0.f;
    for (int i = tid; i < Dz; i += 256) {
        float a = nr[i], c = cr[i];
        d0 += a * wpr[i * 2 + 0] + c * wpr[(Dz + i) * 2 + 0];
        d1 += a * wpr[i * 2 + 1] + c * wpr[(Dz + i) * 2 + 1];
    }
    red[tid] = d0; __syncthreads();
    for (int st = 128; st > 0; st >>= 1) { if (tid < st) red[tid] += red[tid + st]; __syncthreads(); }
    float l0 = red[0] + bpr[0]; __syncthreads();
    red[tid] = d1; __syncthreads();
    for (int st = 128; st > 0; st >>= 1) { if (tid < st) red[tid] += red[tid + st]; __syncthreads(); }
    float l1 = red[0] + bpr[1];
    if (tid == 0) {
        float m = fmaxf(l0, l1);
        float e0 = __expf(l0 - m), e1 = __expf(l1 - m);
        bw[0] = e0 / (e0 + e1); bw[1] = e1 / (e0 + e1);
    }
    __syncthreads();
    // combined scores into LDS
    for (int i = tid; i < NNz; i += 256)
        sc[i] = bw[0] * tsc[(size_t)row * NNz + i] + bw[1] * csc[(size_t)row * NNz + i];
    __syncthreads();
    // iterative top-16 argmax
    for (int t = 0; t < KNz; t++) {
        float bv = -1e30f; int bi = NNz;
        for (int i = tid; i < NNz; i += 256) {
            float v = sc[i];
            if (v > bv || (v == bv && i < bi)) { bv = v; bi = i; }
        }
        rv[tid] = bv; ri[tid] = bi; __syncthreads();
        for (int st = 128; st > 0; st >>= 1) {
            if (tid < st) {
                if (rv[tid + st] > rv[tid] || (rv[tid + st] == rv[tid] && ri[tid + st] < ri[tid])) {
                    rv[tid] = rv[tid + st]; ri[tid] = ri[tid + st];
                }
            }
            __syncthreads();
        }
        if (tid == 0) { bidx[t] = ri[0]; bval[t] = rv[0]; sc[ri[0]] = -1e30f; }
        __syncthreads();
    }
    // softmax over topk (descending -> bval[0] is max)
    if (tid < KNz) wsh[tid] = __expf(bval[tid] - bval[0]);
    __syncthreads();
    float ssum = 0.f;
#pragma unroll
    for (int t = 0; t < KNz; t++) ssum += wsh[t];
    float sinv = 1.0f / ssum;
    // router_out + residual
    for (int i = tid; i < Dz; i += 256) {
        float acc = 0.f;
#pragma unroll
        for (int t = 0; t < KNz; t++) acc += wsh[t] * sinv * neurons[(size_t)bidx[t] * Dz + i];
        rout_f[(size_t)row * Dz + i] = acc;
        rout_h[(size_t)row * Dz + i] = (half_t)acc;
        x1[(size_t)row * Dz + i] = x[(size_t)row * Dz + i] + acc;
    }
    // selection_out + topk_idx
    for (int i = tid; i < NNz; i += 256) sel_out[(size_t)row * NNz + i] = 0.f;
    __syncthreads();
    if (tid < KNz) {
        topk_out[row * KNz + tid] = bidx[tid];
        sel_out[(size_t)row * NNz + bidx[tid]] = wsh[tid] * sinv;
    }
}

// =====================================================================
// pattern selection: pw2 softmax, combine, top-32, softmax weights
// =====================================================================
__global__ __launch_bounds__(256) void k_pffn(const float* __restrict__ n2,
                                              const float* __restrict__ rout,
                                              const float* __restrict__ psc,
                                              const float* __restrict__ rsc,
                                              const float* __restrict__ wpf,
                                              const float* __restrict__ bpf,
                                              float* __restrict__ tkw,
                                              int* __restrict__ tki) {
    int row = blockIdx.x, tid = threadIdx.x;
    __shared__ float red[256];
    __shared__ float sc[NPz];
    __shared__ float rv[256];
    __shared__ int   ri[256];
    __shared__ int   bidx[KPz];
    __shared__ float bval[KPz];
    __shared__ float bw[2];
    const float* nr = n2 + (size_t)row * Dz;
    const float* rr = rout + (size_t)row * Dz;
    float d0 = 0.f, d1 = 0.f;
    for (int i = tid; i < Dz; i += 256) {
        float a = nr[i], c = rr[i];
        d0 += a * wpf[i * 2 + 0] + c * wpf[(Dz + i) * 2 + 0];
        d1 += a * wpf[i * 2 + 1] + c * wpf[(Dz + i) * 2 + 1];
    }
    red[tid] = d0; __syncthreads();
    for (int st = 128; st > 0; st >>= 1) { if (tid < st) red[tid] += red[tid + st]; __syncthreads(); }
    float l0 = red[0] + bpf[0]; __syncthreads();
    red[tid] = d1; __syncthreads();
    for (int st = 128; st > 0; st >>= 1) { if (tid < st) red[tid] += red[tid + st]; __syncthreads(); }
    float l1 = red[0] + bpf[1];
    if (tid == 0) {
        float m = fmaxf(l0, l1);
        float e0 = __expf(l0 - m), e1 = __expf(l1 - m);
        bw[0] = e0 / (e0 + e1); bw[1] = e1 / (e0 + e1);
    }
    __syncthreads();
    for (int i = tid; i < NPz; i += 256)
        sc[i] = bw[0] * psc[(size_t)row * NPz + i] + bw[1] * rsc[(size_t)row * NPz + i];
    __syncthreads();
    for (int t = 0; t < KPz; t++) {
        float bv = -1e30f; int bi = NPz;
        for (int i = tid; i < NPz; i += 256) {
            float v = sc[i];
            if (v > bv || (v == bv && i < bi)) { bv = v; bi = i; }
        }
        rv[tid] = bv; ri[tid] = bi; __syncthreads();
        for (int st = 128; st > 0; st >>= 1) {
            if (tid < st) {
                if (rv[tid + st] > rv[tid] || (rv[tid + st] == rv[tid] && ri[tid + st] < ri[tid])) {
                    rv[tid] = rv[tid + st]; ri[tid] = ri[tid + st];
                }
            }
            __syncthreads();
        }
        if (tid == 0) { bidx[t] = ri[0]; bval[t] = rv[0]; sc[ri[0]] = -1e30f; }
        __syncthreads();
    }
    float ssum = 0.f;
#pragma unroll
    for (int t = 0; t < KPz; t++) ssum += __expf(bval[t] - bval[0]);
    if (tid < KPz) {
        tkw[row * KPz + tid] = __expf(bval[tid] - bval[0]) / ssum;
        tki[row * KPz + tid] = bidx[tid];
    }
}

// =====================================================================
// gate mix + sigmoid-gate + exact GELU -> f16 for down-GEMM
// =====================================================================
__global__ __launch_bounds__(256) void k_gate(const float* __restrict__ hraw,
                                              const float* __restrict__ gates,
                                              const float* __restrict__ tkw,
                                              const int* __restrict__ tki,
                                              half_t* __restrict__ h_h) {
    int row = blockIdx.x, tid = threadIdx.x;
    __shared__ float w[KPz];
    __shared__ int   idx[KPz];
    if (tid < KPz) { w[tid] = tkw[row * KPz + tid]; idx[tid] = tki[row * KPz + tid]; }
    __syncthreads();
    for (int f = tid; f < FFz; f += 256) {
        float g = 0.f;
#pragma unroll
        for (int j = 0; j < KPz; j++) g += w[j] * gates[(size_t)idx[j] * FFz + f];
        float hv = hraw[(size_t)row * FFz + f];
        hv = hv * (1.0f / (1.0f + __expf(-g)));
        float ge = 0.5f * hv * (1.0f + erff(hv * 0.70710678118f));
        h_h[(size_t)row * FFz + f] = (half_t)ge;
    }
}

// =====================================================================
// final residual
// =====================================================================
__global__ void k_resid(const float* __restrict__ x1, const float* __restrict__ f,
                        float* __restrict__ out, int n) {
    for (int i = blockIdx.x * blockDim.x + threadIdx.x; i < n; i += gridDim.x * blockDim.x)
        out[i] = x1[i] + f[i];
}

// =====================================================================
// launch  (workspace requirement ~218 MB; regions reused across phases)
// =====================================================================
extern "C" void kernel_launch(void* const* d_in, const int* in_sizes, int n_in,
                              void* d_out, int out_size, void* d_ws, size_t ws_size,
                              hipStream_t stream) {
    const float* x       = (const float*)d_in[0];
    const float* neurons = (const float*)d_in[1];
    const float* wq  = (const float*)d_in[2];   const float* bq  = (const float*)d_in[3];
    const float* wk  = (const float*)d_in[4];   const float* bk  = (const float*)d_in[5];
    const float* wv  = (const float*)d_in[6];   const float* bv  = (const float*)d_in[7];
    const float* wpr = (const float*)d_in[8];   const float* bpr = (const float*)d_in[9];
    const float* patterns = (const float*)d_in[10];
    const float* gates    = (const float*)d_in[11];
    const float* wpf = (const float*)d_in[12];  const float* bpf = (const float*)d_in[13];
    const float* wup = (const float*)d_in[14];  const float* bup = (const float*)d_in[15];
    const float* wdn = (const float*)d_in[16];  const float* bdn = (const float*)d_in[17];
    const float* g1  = (const float*)d_in[18];  const float* b1  = (const float*)d_in[19];
    const float* g2  = (const float*)d_in[20];  const float* b2  = (const float*)d_in[21];

    float* out_x   = (float*)d_out;                                   // BS*D
    int*   out_idx = (int*)((float*)d_out + (size_t)BSz * Dz);        // BS*KN
    float* out_sel = (float*)d_out + (size_t)BSz * Dz + (size_t)BSz * KNz; // BS*NN

    char* W = (char*)d_ws;
#define MB(x) ((size_t)(x) << 20)
    float*  n1_f  = (float*) (W + MB(0));     // 16 MB
    half_t* n1_h  = (half_t*)(W + MB(16));    //  8
    half_t* q_h   = (half_t*)(W + MB(24));    //  8
    half_t* kk_h  = (half_t*)(W + MB(32));    //  8
    half_t* v_h   = (half_t*)(W + MB(40));    //  8
    half_t* vT_h  = (half_t*)(W + MB(48));    //  8
    float*  ctx_f = (float*) (W + MB(56));    // 16
    half_t* ctx_h = (half_t*)(W + MB(72));    //  8
    float*  tsc   = (float*) (W + MB(80));    // 16
    float*  csc   = (float*) (W + MB(96));    // 16
    half_t* wq_t  = (half_t*)(W + MB(112));   //  2
    half_t* wk_t  = (half_t*)(W + MB(114));   //  2
    half_t* wv_t  = (half_t*)(W + MB(116));   //  2
    half_t* neu_h = (half_t*)(W + MB(118));   //  2
    float*  rou_f = (float*) (W + MB(120));   // 16
    half_t* rou_h = (half_t*)(W + MB(136));   //  8
    float*  x1    = (float*) (W + MB(144));   // 16
    float*  n2_f  = (float*) (W + MB(160));   // 16
    half_t* n2_h  = (half_t*)(W + MB(176));   //  8
    half_t* pat_h = (half_t*)(W + MB(184));   //  1
    float*  psc   = (float*) (W + MB(185));   //  8
    float*  rsc   = (float*) (W + MB(193));   //  8
    float*  tkw   = (float*) (W + MB(201));   // 0.5
    int*    tki   = (int*)   (W + MB(201) + (512u << 10));
    half_t* wup_t = (half_t*)(W + MB(202));   //  8
    half_t* wdn_t = (half_t*)(W + MB(210));   //  8  -> total 218 MB
    // reuse of dead early regions for FFN phase:
    float*  hraw  = (float*) (W + MB(0));     // 64 (over n1..vT/ctx_f, all dead)
    half_t* h_h   = (half_t*)(W + MB(64));    // 32 (over ctx tail/tsc, dead)
    float*  ffo   = (float*) (W + MB(96));    // 16 (over csc, dead)

    const int T = 256;
    // ---- weight conversion ----
    k_cvt_t<<<4096, T, 0, stream>>>(wq,  wq_t,  Dz, Dz);
    k_cvt_t<<<4096, T, 0, stream>>>(wk,  wk_t,  Dz, Dz);
    k_cvt_t<<<4096, T, 0, stream>>>(wv,  wv_t,  Dz, Dz);
    k_cvt_t<<<8192, T, 0, stream>>>(wup, wup_t, Dz, FFz);
    k_cvt_t<<<8192, T, 0, stream>>>(wdn, wdn_t, FFz, Dz);
    k_cvt  <<<4096, T, 0, stream>>>(neurons,  neu_h, NNz * Dz);
    k_cvt  <<<2048, T, 0, stream>>>(patterns, pat_h, NPz * Dz);

    // ---- LN1 + QKV ----
    k_ln<<<BSz, T, 0, stream>>>(x, g1, b1, n1_f, n1_h);
    dim3 gQ(Dz / 64, BSz / 128);
    k_gemm<<<gQ, 128, 0, stream>>>(n1_h, wq_t, bq, nullptr, q_h,  BSz, Dz, Dz);
    k_gemm<<<gQ, 128, 0, stream>>>(n1_h, wk_t, bk, nullptr, kk_h, BSz, Dz, Dz);
    k_gemm<<<gQ, 128, 0, stream>>>(n1_h, wv_t, bv, nullptr, v_h,  BSz, Dz, Dz);
    k_vt<<<8192, T, 0, stream>>>(v_h, vT_h);

    // ---- attention ----
    dim3 gA(Sz / 16, Bz * Hz);
    k_attn<<<gA, 32, 0, stream>>>(q_h, kk_h, vT_h, ctx_f, ctx_h);

    // ---- neuron scores + router ----
    dim3 gN(NNz / 64, BSz / 128);
    k_gemm<<<gN, 128, 0, stream>>>(n1_h,  neu_h, nullptr, tsc, nullptr, BSz, NNz, Dz);
    k_gemm<<<gN, 128, 0, stream>>>(ctx_h, neu_h, nullptr, csc, nullptr, BSz, NNz, Dz);
    k_router<<<BSz, T, 0, stream>>>(x, n1_f, ctx_f, tsc, csc, wpr, bpr, neurons,
                                    rou_f, rou_h, x1, out_idx, out_sel);

    // ---- LN2 + pattern scores + selection ----
    k_ln<<<BSz, T, 0, stream>>>(x1, g2, b2, n2_f, n2_h);
    dim3 gP(NPz / 64, BSz / 128);
    k_gemm<<<gP, 128, 0, stream>>>(n2_h,  pat_h, nullptr, psc, nullptr, BSz, NPz, Dz);
    k_gemm<<<gP, 128, 0, stream>>>(rou_h, pat_h, nullptr, rsc, nullptr, BSz, NPz, Dz);
    k_pffn<<<BSz, T, 0, stream>>>(n2_f, rou_f, psc, rsc, wpf, bpf, tkw, tki);

    // ---- FFN ----
    dim3 gU(FFz / 64, BSz / 128);
    k_gemm<<<gU, 128, 0, stream>>>(n2_h, wup_t, bup, hraw, nullptr, BSz, FFz, Dz);
    k_gate<<<BSz, T, 0, stream>>>(hraw, gates, tkw, tki, h_h);
    dim3 gD(Dz / 64, BSz / 128);
    k_gemm<<<gD, 128, 0, stream>>>(h_h, wdn_t, bdn, ffo, nullptr, BSz, Dz, FFz);
    k_resid<<<8192, T, 0, stream>>>(x1, ffo, out_x, BSz * Dz);
#undef MB
}
